// HGNNLayerDGL_78847009620370
// MI455X (gfx1250) — matrix operand order repaired and checked by reading, not compiled, not run on hardware
//
#include <hip/hip_runtime.h>
#include <math.h>

#define NG 20000
#define ND 8000
#define NS 8000
#define NE 300000
#define DIMC 256
#define NH 8
#define ATT_SCALE 0.17677669529663687f  // 1/sqrt(32)

typedef unsigned short u16;
typedef __attribute__((ext_vector_type(16))) __bf16 bf16x16;
typedef __attribute__((ext_vector_type(8)))  float  f32x8;

__device__ __forceinline__ u16 f2bf(float f) {
  unsigned u = __float_as_uint(f);
  unsigned r = u + 0x7FFFu + ((u >> 16) & 1u);
  return (u16)(r >> 16);
}

__device__ __forceinline__ void atomicMaxF(float* addr, float val) {
  if (val >= 0.f) atomicMax((int*)addr, __float_as_int(val));
  else            atomicMin((unsigned int*)addr, (unsigned int)__float_as_int(val));
}

// ---------------------------------------------------------------------------
// f32 -> bf16 conversion (grid-stride)
// ---------------------------------------------------------------------------
__global__ __launch_bounds__(256) void cvt_bf16_kernel(const float* __restrict__ x,
                                                       u16* __restrict__ y, int n) {
  for (int i = blockIdx.x * blockDim.x + threadIdx.x; i < n;
       i += gridDim.x * blockDim.x)
    y[i] = f2bf(x[i]);
}

__global__ __launch_bounds__(256) void fill_kernel(float* __restrict__ x, float v, int n) {
  int i = blockIdx.x * blockDim.x + threadIdx.x;
  if (i < n) x[i] = v;
}

__global__ __launch_bounds__(256) void flag_kernel(const int* __restrict__ dst,
                                                   int* __restrict__ flag, int n) {
  int i = blockIdx.x * blockDim.x + threadIdx.x;
  if (i < n) flag[dst[i]] = 1;
}

// ---------------------------------------------------------------------------
// WMMA GEMM: Y[rows,256] = X[rows,256](bf16) * W^T (W[256,256] bf16, row-major)
//            + bias, optional tanh epilogue, f32 output.
// Wave tile: 32(M) x 64(N)  -> 2 A frags x 4 B frags = 8 WMMAs / K-step.
// Software-pipelined: K-step k+32 fragments are loaded before the WMMAs of
// step k issue, so loads stay in flight behind the matrix pipe.
// Block: 8 waves -> 256(M) x 64(N) block tile.
// ---------------------------------------------------------------------------
struct Frags {
  bf16x16 a0, a1;
  bf16x16 b0, b1, b2, b3;
};

__device__ __forceinline__ bf16x16 ld_frag(const u16* p0, const u16* p1) {
  union { uint4 u[2]; bf16x16 v; } f;
  f.u[0] = *(const uint4*)p0;
  f.u[1] = *(const uint4*)p1;
  return f.v;
}

__global__ __launch_bounds__(256) void gemm_bf16_kernel(
    const u16* __restrict__ X, const u16* __restrict__ W,
    const float* __restrict__ bias, float* __restrict__ Y,
    int nrows, int dotanh) {
  const int lane  = threadIdx.x & 31;
  const int wave  = threadIdx.x >> 5;
  const int lmod  = lane & 15;
  const int lhalf = lane >> 4;
  const int rowBase = blockIdx.x * 256 + wave * 32;
  const int colBase = blockIdx.y * 64;

  int ar0 = rowBase + lmod;        // rows 0..15 of wave tile
  int ar1 = rowBase + 16 + lmod;   // rows 16..31 of wave tile
  if (ar0 >= nrows) ar0 = nrows - 1;  // clamp: keep EXEC all-1s for WMMA
  if (ar1 >= nrows) ar1 = nrows - 1;
  const u16* Xr0 = X + (size_t)ar0 * DIMC;
  const u16* Xr1 = X + (size_t)ar1 * DIMC;

  const u16* Wr[4];
#pragma unroll
  for (int t = 0; t < 4; ++t)
    Wr[t] = W + (size_t)(colBase + t * 16 + lmod) * DIMC;

  auto loadk = [&](int k) -> Frags {
    Frags f;
    f.a0 = ld_frag(Xr0 + k + lhalf * 16, Xr0 + k + lhalf * 16 + 8);
    f.a1 = ld_frag(Xr1 + k + lhalf * 16, Xr1 + k + lhalf * 16 + 8);
    f.b0 = ld_frag(Wr[0] + k + lhalf * 16, Wr[0] + k + lhalf * 16 + 8);
    f.b1 = ld_frag(Wr[1] + k + lhalf * 16, Wr[1] + k + lhalf * 16 + 8);
    f.b2 = ld_frag(Wr[2] + k + lhalf * 16, Wr[2] + k + lhalf * 16 + 8);
    f.b3 = ld_frag(Wr[3] + k + lhalf * 16, Wr[3] + k + lhalf * 16 + 8);
    return f;
  };

  f32x8 zero = {0.f, 0.f, 0.f, 0.f, 0.f, 0.f, 0.f, 0.f};
  f32x8 accA[4] = {zero, zero, zero, zero};
  f32x8 accB[4] = {zero, zero, zero, zero};

  Frags cur = loadk(0);
#pragma unroll
  for (int k = 0; k < DIMC; k += 32) {
    if (k + 64 < DIMC) {
      __builtin_prefetch(Xr0 + k + 64, 0, 3);  // global_prefetch_b8
      __builtin_prefetch(Xr1 + k + 64, 0, 3);
    }
    Frags nxt = (k + 32 < DIMC) ? loadk(k + 32) : cur;
    accA[0] = __builtin_amdgcn_wmma_f32_16x16x32_bf16(false, cur.a0, false, cur.b0, (short)0, accA[0], false, false);
    accB[0] = __builtin_amdgcn_wmma_f32_16x16x32_bf16(false, cur.a1, false, cur.b0, (short)0, accB[0], false, false);
    accA[1] = __builtin_amdgcn_wmma_f32_16x16x32_bf16(false, cur.a0, false, cur.b1, (short)0, accA[1], false, false);
    accB[1] = __builtin_amdgcn_wmma_f32_16x16x32_bf16(false, cur.a1, false, cur.b1, (short)0, accB[1], false, false);
    accA[2] = __builtin_amdgcn_wmma_f32_16x16x32_bf16(false, cur.a0, false, cur.b2, (short)0, accA[2], false, false);
    accB[2] = __builtin_amdgcn_wmma_f32_16x16x32_bf16(false, cur.a1, false, cur.b2, (short)0, accB[2], false, false);
    accA[3] = __builtin_amdgcn_wmma_f32_16x16x32_bf16(false, cur.a0, false, cur.b3, (short)0, accA[3], false, false);
    accB[3] = __builtin_amdgcn_wmma_f32_16x16x32_bf16(false, cur.a1, false, cur.b3, (short)0, accB[3], false, false);
    cur = nxt;
  }

#pragma unroll
  for (int t = 0; t < 4; ++t) {
    int n = colBase + t * 16 + lmod;
    float bs = bias[n];
#pragma unroll
    for (int r = 0; r < 8; ++r) {
      int m0 = rowBase + lhalf * 8 + r;
      if (m0 < nrows) {
        float v = accA[t][r] + bs;
        if (dotanh) v = tanhf(v);
        Y[(size_t)m0 * DIMC + n] = v;
      }
      int m1 = m0 + 16;
      if (m1 < nrows) {
        float v = accB[t][r] + bs;
        if (dotanh) v = tanhf(v);
        Y[(size_t)m1 * DIMC + n] = v;
      }
    }
  }
}

// ---------------------------------------------------------------------------
// Edge attention: one wave32 per edge; lane handles 8 dims; head = lane/4.
// ---------------------------------------------------------------------------
__global__ __launch_bounds__(256) void edge_score_kernel(
    const float* __restrict__ K, const float* __restrict__ Q,
    const int* __restrict__ src, const int* __restrict__ dst,
    float* __restrict__ score, float* __restrict__ mmax, int nedge) {
  int e = blockIdx.x * 8 + (threadIdx.x >> 5);
  if (e >= nedge) return;
  int lane = threadIdx.x & 31;
  int s = src[e], d = dst[e];
  const float4* kp = (const float4*)(K + (size_t)s * DIMC + lane * 8);
  const float4* qp = (const float4*)(Q + (size_t)d * DIMC + lane * 8);
  float4 k0 = kp[0], k1 = kp[1], q0 = qp[0], q1 = qp[1];
  float p = k0.x * q0.x + k0.y * q0.y + k0.z * q0.z + k0.w * q0.w +
            k1.x * q1.x + k1.y * q1.y + k1.z * q1.z + k1.w * q1.w;
  p += __shfl_xor(p, 1);
  p += __shfl_xor(p, 2);
  if ((lane & 3) == 0) {
    float sc = p * ATT_SCALE;
    int h = lane >> 2;
    score[(size_t)e * NH + h] = sc;
    atomicMaxF(&mmax[(size_t)d * NH + h], sc);
  }
}

__global__ __launch_bounds__(256) void edge_exp_kernel(
    const float* __restrict__ mmax, const int* __restrict__ dst,
    float* __restrict__ a, float* __restrict__ den, int nedge) {
  int i = blockIdx.x * blockDim.x + threadIdx.x;
  if (i >= nedge * NH) return;
  int e = i >> 3, h = i & 7;
  int d = dst[e];
  float v = expf(a[i] - mmax[(size_t)d * NH + h]);
  a[i] = v;
  atomicAdd(&den[(size_t)d * NH + h], v);
}

__global__ __launch_bounds__(256) void edge_accum_kernel(
    const float* __restrict__ V, const float* __restrict__ a,
    const float* __restrict__ den, const int* __restrict__ src,
    const int* __restrict__ dst, float* __restrict__ out, int nedge) {
  int e = blockIdx.x * 8 + (threadIdx.x >> 5);
  if (e >= nedge) return;
  int lane = threadIdx.x & 31;
  int s = src[e], d = dst[e];
  int h = lane >> 2;
  float w = a[(size_t)e * NH + h] / den[(size_t)d * NH + h];
  const float* vp = V + (size_t)s * DIMC + lane * 8;
  float* op = out + (size_t)d * DIMC + lane * 8;
#pragma unroll
  for (int j = 0; j < 8; ++j) atomicAdd(op + j, vp[j] * w);
}

// ---------------------------------------------------------------------------
// Gene fusion + GELU + residual + LayerNorm (one wave32 per row)
// ---------------------------------------------------------------------------
__global__ __launch_bounds__(256) void gene_finalize_kernel(
    const float* __restrict__ gmd, const float* __restrict__ gms,
    const float* __restrict__ T1, const float* __restrict__ T2,
    const float* __restrict__ fv, const float* __restrict__ fbv,
    const int* __restrict__ hasd, const int* __restrict__ hass,
    const float* __restrict__ feat,
    const float* __restrict__ lng, const float* __restrict__ lnb,
    float* __restrict__ out, u16* __restrict__ out_bf, int n) {
  int row = blockIdx.x * 8 + (threadIdx.x >> 5);
  if (row >= n) return;
  int lane = threadIdx.x & 31;
  int j0 = lane * 8;
  size_t base = (size_t)row * DIMC + j0;

  float p1 = 0.f, p2 = 0.f;
#pragma unroll
  for (int j = 0; j < 8; ++j) {
    float vv = fv[j0 + j];
    p1 += T1[base + j] * vv;
    p2 += T2[base + j] * vv;
  }
#pragma unroll
  for (int o = 1; o < 32; o <<= 1) {
    p1 += __shfl_xor(p1, o);
    p2 += __shfl_xor(p2, o);
  }
  float e1 = p1 + fbv[0], e2 = p2 + fbv[0];
  float mx = fmaxf(e1, e2);
  float x1 = expf(e1 - mx), x2 = expf(e2 - mx);
  float w1 = x1 / (x1 + x2);
  float w2 = 1.f - w1;
  int hd = hasd[row], hs = hass[row];

  float x[8];
#pragma unroll
  for (int j = 0; j < 8; ++j) {
    float md = gmd[base + j], ms = gms[base + j];
    float msg = (hd && hs) ? (w1 * md + w2 * ms) : (hd ? md : (hs ? ms : 0.f));
    float gel = 0.5f * msg * (1.f + erff(msg * 0.70710678118654752f));
    x[j] = feat[base + j] + gel;
  }
  float s = 0.f;
#pragma unroll
  for (int j = 0; j < 8; ++j) s += x[j];
#pragma unroll
  for (int o = 1; o < 32; o <<= 1) s += __shfl_xor(s, o);
  float mu = s * (1.f / 256.f);
  float vs = 0.f;
#pragma unroll
  for (int j = 0; j < 8; ++j) { float d = x[j] - mu; vs += d * d; }
#pragma unroll
  for (int o = 1; o < 32; o <<= 1) vs += __shfl_xor(vs, o);
  float inv = rsqrtf(vs * (1.f / 256.f) + 1e-5f);
#pragma unroll
  for (int j = 0; j < 8; ++j) {
    float y = (x[j] - mu) * inv * lng[j0 + j] + lnb[j0 + j];
    out[base + j] = y;
    out_bf[base + j] = f2bf(y);
  }
}

__global__ __launch_bounds__(256) void node_finalize_kernel(
    const float* __restrict__ msg, const float* __restrict__ feat,
    const float* __restrict__ lng, const float* __restrict__ lnb,
    float* __restrict__ out, int n) {
  int row = blockIdx.x * 8 + (threadIdx.x >> 5);
  if (row >= n) return;
  int lane = threadIdx.x & 31;
  int j0 = lane * 8;
  size_t base = (size_t)row * DIMC + j0;
  float x[8];
#pragma unroll
  for (int j = 0; j < 8; ++j) {
    float mm = msg[base + j];
    float gel = 0.5f * mm * (1.f + erff(mm * 0.70710678118654752f));
    x[j] = feat[base + j] + gel;
  }
  float s = 0.f;
#pragma unroll
  for (int j = 0; j < 8; ++j) s += x[j];
#pragma unroll
  for (int o = 1; o < 32; o <<= 1) s += __shfl_xor(s, o);
  float mu = s * (1.f / 256.f);
  float vs = 0.f;
#pragma unroll
  for (int j = 0; j < 8; ++j) { float d = x[j] - mu; vs += d * d; }
#pragma unroll
  for (int o = 1; o < 32; o <<= 1) vs += __shfl_xor(vs, o);
  float inv = rsqrtf(vs * (1.f / 256.f) + 1e-5f);
#pragma unroll
  for (int j = 0; j < 8; ++j)
    out[base + j] = (x[j] - mu) * inv * lng[j0 + j] + lnb[j0 + j];
}

// ---------------------------------------------------------------------------
// Host orchestration
// ---------------------------------------------------------------------------
extern "C" void kernel_launch(void* const* d_in, const int* in_sizes, int n_in,
                              void* d_out, int out_size, void* d_ws, size_t ws_size,
                              hipStream_t stream) {
  (void)in_sizes; (void)n_in; (void)out_size; (void)ws_size;
  const float* gene_feat = (const float*)d_in[0];
  const float* drug_feat = (const float*)d_in[1];
  const float* dis_feat  = (const float*)d_in[2];
  const float* W_f32[14] = {
      (const float*)d_in[3],  (const float*)d_in[5],  (const float*)d_in[7],
      (const float*)d_in[9],  (const float*)d_in[11], (const float*)d_in[13],
      (const float*)d_in[15], (const float*)d_in[17], (const float*)d_in[19],
      (const float*)d_in[21], (const float*)d_in[23], (const float*)d_in[25],
      (const float*)d_in[27], (const float*)d_in[29]};
  const float* dg_bq   = (const float*)d_in[4];
  const float* dg_bk   = (const float*)d_in[6];
  const float* dg_bv   = (const float*)d_in[8];
  const float* disg_bq = (const float*)d_in[10];
  const float* disg_bk = (const float*)d_in[12];
  const float* disg_bv = (const float*)d_in[14];
  const float* gd_bq   = (const float*)d_in[16];
  const float* gd_bk   = (const float*)d_in[18];
  const float* gd_bv   = (const float*)d_in[20];
  const float* gdis_bq = (const float*)d_in[22];
  const float* gdis_bk = (const float*)d_in[24];
  const float* gdis_bv = (const float*)d_in[26];
  const float* f_b1 = (const float*)d_in[28];
  const float* f_b2 = (const float*)d_in[30];
  const float* f_v  = (const float*)d_in[31];
  const float* f_bv = (const float*)d_in[32];
  const float* ln1_g = (const float*)d_in[33];
  const float* ln1_b = (const float*)d_in[34];
  const float* ln2_g = (const float*)d_in[35];
  const float* ln2_b = (const float*)d_in[36];
  const int* dg_src   = (const int*)d_in[37];
  const int* dg_dst   = (const int*)d_in[38];
  const int* disg_src = (const int*)d_in[39];
  const int* disg_dst = (const int*)d_in[40];
  const int* gd_src   = (const int*)d_in[41];
  const int* gd_dst   = (const int*)d_in[42];
  const int* gdis_src = (const int*)d_in[43];
  const int* gdis_dst = (const int*)d_in[44];

  char* wp = (char*)d_ws;
  auto alloc = [&](size_t bytes) -> void* {
    void* p = (void*)wp;
    wp += (bytes + 255) & ~(size_t)255;
    return p;
  };
  const size_t NGD = (size_t)NG * DIMC, NDD = (size_t)ND * DIMC, NSD = (size_t)NS * DIMC;

  u16* gene_bf = (u16*)alloc(NGD * 2);
  u16* drug_bf = (u16*)alloc(NDD * 2);
  u16* dis_bf  = (u16*)alloc(NSD * 2);
  u16* go_bf   = (u16*)alloc(NGD * 2);
  u16* gmd_bf  = (u16*)alloc(NGD * 2);
  u16* gms_bf  = (u16*)alloc(NGD * 2);
  u16* Wbf[14];
  for (int i = 0; i < 14; ++i) Wbf[i] = (u16*)alloc((size_t)DIMC * DIMC * 2);

  float* bufNG1 = (float*)alloc(NGD * 4);  // Qg(dg) -> T1 -> Kg(gdis)
  float* bufNG2 = (float*)alloc(NGD * 4);  // Qg(disg) -> T2 -> Vg(gdis)
  float* bufNG3 = (float*)alloc(NGD * 4);  // gm_drug -> Kg(gd)
  float* bufNG4 = (float*)alloc(NGD * 4);  // gm_dis  -> Vg(gd)
  float* bufD1  = (float*)alloc(NDD * 4);  // Kd -> Qd
  float* bufD2  = (float*)alloc(NDD * 4);  // Vd -> drug_msg
  float* bufS1  = (float*)alloc(NSD * 4);  // Ks -> Qs
  float* bufS2  = (float*)alloc(NSD * 4);  // Vs -> dis_msg
  float* abuf   = (float*)alloc((size_t)NE * NH * 4);
  float* mbuf   = (float*)alloc((size_t)NG * NH * 4);
  float* den    = (float*)alloc((size_t)NG * NH * 4);
  int*   hasd   = (int*)alloc((size_t)NG * 4);
  int*   hass   = (int*)alloc((size_t)NG * 4);

  float* out_drug = (float*)d_out;
  float* out_dis  = (float*)d_out + NDD;
  float* out_gene = (float*)d_out + NDD + NSD;

  auto cvt = [&](const float* x, u16* y, size_t nelem) {
    int n = (int)nelem;
    int nb = (n + 1023) / 1024;
    if (nb > 16384) nb = 16384;
    cvt_bf16_kernel<<<nb, 256, 0, stream>>>(x, y, n);
  };
  auto gemm = [&](const u16* X, const u16* W, const float* bias, float* Y,
                  int rows, int dotanh) {
    dim3 g((rows + 255) / 256, 4);
    gemm_bf16_kernel<<<g, dim3(256), 0, stream>>>(X, W, bias, Y, rows, dotanh);
  };
  auto relation = [&](const float* Kmat, const float* Qmat, const float* Vmat,
                      const int* src, const int* dst, float* outv, int ndst) {
    fill_kernel<<<((ndst * NH) + 255) / 256, 256, 0, stream>>>(
        mbuf, -__builtin_huge_valf(), ndst * NH);
    hipMemsetAsync(den, 0, (size_t)ndst * NH * 4, stream);
    hipMemsetAsync(outv, 0, (size_t)ndst * DIMC * 4, stream);
    edge_score_kernel<<<(NE + 7) / 8, 256, 0, stream>>>(Kmat, Qmat, src, dst, abuf, mbuf, NE);
    edge_exp_kernel<<<((NE * NH) + 255) / 256, 256, 0, stream>>>(mbuf, dst, abuf, den, NE);
    edge_accum_kernel<<<(NE + 7) / 8, 256, 0, stream>>>(Vmat, abuf, den, src, dst, outv, NE);
  };

  // --- prep: bf16 copies of features and 14 weight matrices ---
  cvt(gene_feat, gene_bf, NGD);
  cvt(drug_feat, drug_bf, NDD);
  cvt(dis_feat, dis_bf, NSD);
  for (int i = 0; i < 14; ++i) cvt(W_f32[i], Wbf[i], (size_t)DIMC * DIMC);

  // --- phase 1: drug->gene and disease->gene attention ---
  gemm(gene_bf, Wbf[0], dg_bq, bufNG1, NG, 0);    // Q genes (dg)
  gemm(drug_bf, Wbf[1], dg_bk, bufD1, ND, 0);     // K drugs
  gemm(drug_bf, Wbf[2], dg_bv, bufD2, ND, 0);     // V drugs
  gemm(gene_bf, Wbf[3], disg_bq, bufNG2, NG, 0);  // Q genes (disg)
  gemm(dis_bf,  Wbf[4], disg_bk, bufS1, NS, 0);   // K diseases
  gemm(dis_bf,  Wbf[5], disg_bv, bufS2, NS, 0);   // V diseases

  relation(bufD1, bufNG1, bufD2, dg_src, dg_dst, bufNG3, NG);      // gm_drug
  relation(bufS1, bufNG2, bufS2, disg_src, disg_dst, bufNG4, NG);  // gm_dis

  hipMemsetAsync(hasd, 0, (size_t)NG * 4, stream);
  hipMemsetAsync(hass, 0, (size_t)NG * 4, stream);
  flag_kernel<<<(NE + 255) / 256, 256, 0, stream>>>(dg_dst, hasd, NE);
  flag_kernel<<<(NE + 255) / 256, 256, 0, stream>>>(disg_dst, hass, NE);

  // --- Bahdanau fusion GEMMs (tanh epilogue) ---
  cvt(bufNG3, gmd_bf, NGD);
  cvt(bufNG4, gms_bf, NGD);
  gemm(gmd_bf, Wbf[12], f_b1, bufNG1, NG, 1);  // T1
  gemm(gms_bf, Wbf[13], f_b2, bufNG2, NG, 1);  // T2

  gene_finalize_kernel<<<(NG + 7) / 8, 256, 0, stream>>>(
      bufNG3, bufNG4, bufNG1, bufNG2, f_v, f_bv, hasd, hass,
      gene_feat, ln1_g, ln1_b, out_gene, go_bf, NG);

  // --- phase 2: gene->drug and gene->disease attention ---
  gemm(drug_bf, Wbf[6], gd_bq, bufD1, ND, 0);      // Q drugs
  gemm(go_bf,   Wbf[7], gd_bk, bufNG3, NG, 0);     // K genes (gd)
  gemm(go_bf,   Wbf[8], gd_bv, bufNG4, NG, 0);     // V genes (gd)
  gemm(dis_bf,  Wbf[9], gdis_bq, bufS1, NS, 0);    // Q diseases
  gemm(go_bf,   Wbf[10], gdis_bk, bufNG1, NG, 0);  // K genes (gdis)
  gemm(go_bf,   Wbf[11], gdis_bv, bufNG2, NG, 0);  // V genes (gdis)

  relation(bufNG3, bufD1, bufNG4, gd_src, gd_dst, bufD2, ND);      // drug_msg
  relation(bufNG1, bufS1, bufNG2, gdis_src, gdis_dst, bufS2, NS);  // dis_msg

  node_finalize_kernel<<<(ND + 7) / 8, 256, 0, stream>>>(
      bufD2, drug_feat, ln2_g, ln2_b, out_drug, ND);
  node_finalize_kernel<<<(NS + 7) / 8, 256, 0, stream>>>(
      bufS2, dis_feat, ln2_g, ln2_b, out_dis, NS);
}
